// Transformer_23450521436954
// MI455X (gfx1250) — matrix operand order repaired and checked by reading, not compiled, not run on hardware
//
#include <hip/hip_runtime.h>
#include <hip/hip_bf16.h>

// ---------------- problem constants ----------------
#define NSEQ 4096
#define DIMD 1536
#define NH   8
#define DKK  64
#define DVV  192
#define NPOS 8191  // 2N-1

typedef __attribute__((ext_vector_type(16))) _Float16     v16h;
typedef __attribute__((ext_vector_type(8)))  float        v8f;
typedef __attribute__((ext_vector_type(4)))  unsigned int v4u;

union Frag16 { v16h v; v4u q[2]; _Float16 h[16]; };
union H8     { v4u q; _Float16 h[8]; };

// ---- async global->LDS staging (gfx1250 GLOBAL_LOAD_ASYNC_TO_LDS_B128) ----
#if defined(__gfx1250__) &&                                         \
    __has_builtin(__builtin_amdgcn_global_load_async_to_lds_b128) && \
    __has_builtin(__builtin_amdgcn_s_wait_asynccnt)
#define USE_ASYNC 1
#else
#define USE_ASYNC 0
#endif

#if USE_ASYNC
// Builtin expects pointers to GCC-style int vectors (vector_size(16)):
//   arg0: global (AS1) int4*, arg1: LDS (AS3) int4*, arg2/3: imm offset / cpol.
typedef int v4i_t __attribute__((vector_size(16)));
typedef __attribute__((address_space(1))) v4i_t GV4;
typedef __attribute__((address_space(3))) v4i_t LV4;
__device__ __forceinline__ void async_cp16(const _Float16* g, _Float16* l) {
  __builtin_amdgcn_global_load_async_to_lds_b128((GV4*)g, (LV4*)l, 0, 0);
}
__device__ __forceinline__ void async_wait0() {
  __builtin_amdgcn_s_wait_asynccnt(0);
}
#else
__device__ __forceinline__ void async_cp16(const _Float16* g, _Float16* l) {
  *(v4u*)l = *(const v4u*)g;
}
__device__ __forceinline__ void async_wait0() {}
#endif

// ---- WMMA fragment loaders (wave32; contiguous k = 16*hi + e, same for A and B) ----
// A (16x32 f16): lane m = lane&15; element e -> k = e + 16*hi
//   => one contiguous 16-half run at k0+16*hi (matches B).
__device__ __forceinline__ v16h ld_frag_a(const _Float16* base, int pitch, int row,
                                          int k0, int hi) {
  Frag16 f;
  const _Float16* r = base + row * pitch + k0;
  f.q[0] = *(const v4u*)(r + 16 * hi);
  f.q[1] = *(const v4u*)(r + 16 * hi + 8);
  return f.v;
}
// B (32x16 f16): lane n = lane&15; element e -> k = e + 16*hi
//   => one contiguous 16-half run; requires B stored [n][k] in LDS.
__device__ __forceinline__ v16h ld_frag_b(const _Float16* base, int pitch, int ncol,
                                          int k0, int hi) {
  Frag16 f;
  const _Float16* r = base + ncol * pitch + k0 + 16 * hi;
  f.q[0] = *(const v4u*)(r);
  f.q[1] = *(const v4u*)(r + 8);
  return f.v;
}
__device__ __forceinline__ v8f wmma16(v16h a, v16h b, v8f c) {
  return __builtin_amdgcn_wmma_f32_16x16x32_f16(false, a, false, b, (short)0, c,
                                                false, false);
}

// ---------------- LayerNorm (eps=1e-3) + f16 cast ----------------
__global__ void __launch_bounds__(256) ln_k(const float* __restrict__ x,
                                            const float* __restrict__ gam,
                                            const float* __restrict__ bet,
                                            _Float16* __restrict__ xn) {
  __shared__ float red[256];
  const int row = blockIdx.x, t = threadIdx.x;
  const float* xr = x + (size_t)row * DIMD;
  float s = 0.f;
  for (int i = t; i < DIMD; i += 256) s += xr[i];
  red[t] = s; __syncthreads();
  for (int o = 128; o > 0; o >>= 1) { if (t < o) red[t] += red[t + o]; __syncthreads(); }
  float mu = red[0] * (1.0f / DIMD);
  __syncthreads();
  float v = 0.f;
  for (int i = t; i < DIMD; i += 256) { float d = xr[i] - mu; v += d * d; }
  red[t] = v; __syncthreads();
  for (int o = 128; o > 0; o >>= 1) { if (t < o) red[t] += red[t + o]; __syncthreads(); }
  float rs = rsqrtf(red[0] * (1.0f / DIMD) + 1e-3f);
  for (int i = t; i < DIMD; i += 256)
    xn[(size_t)row * DIMD + i] = (_Float16)((xr[i] - mu) * rs * gam[i] + bet[i]);
}

// ---------------- f32 -> f16 cast ----------------
__global__ void __launch_bounds__(256) cast_k(const float* __restrict__ in,
                                              _Float16* __restrict__ out, int n) {
  int i = blockIdx.x * 256 + threadIdx.x;
  if (i < n) out[i] = (_Float16)in[i];
}

// -------- positional basis @ Wr -> per-head rel_k (f16, [H][8191][64]) --------
__global__ void __launch_bounds__(128) relk_k(const float* __restrict__ Wr,
                                              _Float16* __restrict__ RK) {
  __shared__ float feat[32];
  const int p = blockIdx.x;                 // 0..8190
  const float d = (float)(p - (NSEQ - 1));  // distance
  if (threadIdx.x < 32) {
    int f = threadIdx.x, ff = f & 15;
    float pr = __expf(__logf((float)NSEQ + 1.0f) / 16.0f);  // pow_rate
    float cw = __powf(pr, (float)(ff + 1));                  // center width
    float e = (cw > fabsf(d)) ? 1.0f : 0.0f;
    if (f >= 16) e *= (d > 0.f) ? 1.f : ((d < 0.f) ? -1.f : 0.f);  // sign(d)*emb
    feat[f] = e;
  }
  __syncthreads();
  for (int c = threadIdx.x * 4; c < threadIdx.x * 4 + 4; ++c) {
    float s = 0.f;
    #pragma unroll
    for (int f = 0; f < 32; ++f) s += feat[f] * Wr[f * (NH * DKK) + c];
    int h = c >> 6, dk = c & 63;
    RK[((size_t)h * NPOS + p) * DKK + dk] = (_Float16)s;
  }
}

// -------- Q epilogue: scale + content/positional bias -> two f16 copies --------
__global__ void __launch_bounds__(256) qbias_k(const float* __restrict__ Q,
                                               const float* __restrict__ cb,
                                               const float* __restrict__ pb,
                                               _Float16* __restrict__ Qc,
                                               _Float16* __restrict__ Qp) {
  int i = blockIdx.x * 256 + threadIdx.x;  // over 4096*512
  int col = i & 511;
  float qs = Q[i] * 0.125f;  // DK^-0.5
  Qc[i] = (_Float16)(qs + cb[col]);
  Qp[i] = (_Float16)(qs + pb[col]);
}

// ---------------- generic 64x64 WMMA GEMM, BK=32, 8 waves ----------------
// C[M x Nc] = A[M x K] * B[K x Nc]; F32OUT=1 -> f32 out (+bias), else f16 out.
template <int F32OUT>
__global__ void __launch_bounds__(256) gemm_k(const _Float16* __restrict__ A,
                                              const _Float16* __restrict__ B,
                                              int Nc, int K,
                                              _Float16* __restrict__ C16,
                                              float* __restrict__ C32,
                                              const float* __restrict__ bias) {
  __shared__ __align__(16) _Float16 As[64 * 40];   // [m][k], pitch 40
  __shared__ __align__(16) _Float16 Bst[64 * 40];  // transposed [n][k], pitch 40
  const int t = threadIdx.x;
  const int lane = t & 31, w = t >> 5, hi = lane >> 4, ln = lane & 15;
  const int m0 = blockIdx.x * 64, n0 = blockIdx.y * 64;
  const int rt = w >> 1, cp = w & 1;  // wave -> (row tile, col half)

  v8f acc0 = {0.f, 0.f, 0.f, 0.f, 0.f, 0.f, 0.f, 0.f};
  v8f acc1 = acc0;

  const int arow = t >> 2, acb = t & 3;   // A-tile copy assignment
  const int bkr = t >> 3, bcb = t & 7;    // B-tile copy assignment

  for (int k0 = 0; k0 < K; k0 += 32) {
    __syncthreads();
    // A tile 64x32: straight 16B copies -> async global->LDS path
    async_cp16(A + (size_t)(m0 + arow) * K + k0 + 8 * acb, As + arow * 40 + 8 * acb);
    // prefetch next chunks into cache (global_prefetch_b8)
    if (k0 + 32 < K) {
      __builtin_prefetch(A + (size_t)(m0 + arow) * K + k0 + 32 + 8 * acb, 0, 1);
      __builtin_prefetch(B + (size_t)(k0 + 32 + bkr) * Nc + n0 + 8 * bcb, 0, 1);
    }
    {  // B tile 32x64, scattered into transposed LDS [n][k]
      H8 tb;
      tb.q = *(const v4u*)(B + (size_t)(k0 + bkr) * Nc + n0 + 8 * bcb);
      #pragma unroll
      for (int u = 0; u < 8; ++u) Bst[(8 * bcb + u) * 40 + bkr] = tb.h[u];
    }
    async_wait0();
    __syncthreads();
    v16h a  = ld_frag_a(As, 40, rt * 16 + ln, 0, hi);
    v16h b0 = ld_frag_b(Bst, 40, cp * 32 + ln, 0, hi);
    v16h b1 = ld_frag_b(Bst, 40, cp * 32 + 16 + ln, 0, hi);
    acc0 = wmma16(a, b0, acc0);
    acc1 = wmma16(a, b1, acc1);
  }
  #pragma unroll
  for (int r = 0; r < 8; ++r) {  // C/D layout: row = r + 8*hi, col = lane&15
    int row = m0 + rt * 16 + r + 8 * hi;
    int c0 = n0 + cp * 32 + ln, c1 = c0 + 16;
    if (F32OUT) {
      float b0v = bias ? bias[c0] : 0.f;
      float b1v = bias ? bias[c1] : 0.f;
      C32[(size_t)row * Nc + c0] = acc0[r] + b0v;
      C32[(size_t)row * Nc + c1] = acc1[r] + b1v;
    } else {
      C16[(size_t)row * Nc + c0] = (_Float16)acc0[r];
      C16[(size_t)row * Nc + c1] = (_Float16)acc1[r];
    }
  }
}

// ---------------- flash attention with relative-shift band GEMM ----------------
// grid (N/64, H), 256 threads. Dynamic LDS ~132KB (WGP has 320KB).
#define AT_QC 0
#define AT_QP 9216
#define AT_KS 18432
#define AT_VS 27648   // transposed [dv][key], 192 x 72 halves
#define AT_RS 55296   // rel band, 128 x 72 halves
#define AT_PS 73728   // P (f16), 64 x 72
#define AT_SF 82944   // S (f32), 64 x 68
#define AT_S2 100352  // band logits (f32), 64 x 132
#define AT_ST 134144  // m / l / alpha, 3 x 64 f32
#define AT_SMEM 134912

__global__ void __launch_bounds__(256) attn_k(const _Float16* __restrict__ Qc,
                                              const _Float16* __restrict__ Qp,
                                              const _Float16* __restrict__ Kp,
                                              const _Float16* __restrict__ Vp,
                                              const _Float16* __restrict__ Rk,
                                              _Float16* __restrict__ Oo) {
  extern __shared__ __align__(16) char smem[];
  _Float16* QCs = (_Float16*)(smem + AT_QC);
  _Float16* QPs = (_Float16*)(smem + AT_QP);
  _Float16* KSs = (_Float16*)(smem + AT_KS);
  _Float16* VSs = (_Float16*)(smem + AT_VS);
  _Float16* RSs = (_Float16*)(smem + AT_RS);
  _Float16* PSs = (_Float16*)(smem + AT_PS);
  float* Sf = (float*)(smem + AT_SF);
  float* S2 = (float*)(smem + AT_S2);
  float* mrow = (float*)(smem + AT_ST);
  float* lrow = mrow + 64;
  float* arow = lrow + 64;

  const int t = threadIdx.x;
  const int lane = t & 31, w = t >> 5, hi = lane >> 4, ln = lane & 15;
  const int h = blockIdx.y;
  const int i0 = blockIdx.x * 64;
  const int rt = w >> 1, cw = w & 1;
  const v8f zero8 = {0.f, 0.f, 0.f, 0.f, 0.f, 0.f, 0.f, 0.f};

  {  // load biased query tiles (64 x 64 each) via async copy
    int row = t >> 2, cb = t & 3;
    size_t g = (size_t)(i0 + row) * 512 + h * 64 + cb * 16;
    async_cp16(Qc + g,     QCs + row * 72 + cb * 16);
    async_cp16(Qc + g + 8, QCs + row * 72 + cb * 16 + 8);
    async_cp16(Qp + g,     QPs + row * 72 + cb * 16);
    async_cp16(Qp + g + 8, QPs + row * 72 + cb * 16 + 8);
  }
  if (t < 64) { mrow[t] = -3.0e38f; lrow[t] = 0.f; }
  async_wait0();

  v8f o[6];
  #pragma unroll
  for (int i = 0; i < 6; ++i) o[i] = zero8;

  for (int j0 = 0; j0 < NSEQ; j0 += 64) {
    __syncthreads();  // prior PV done reading PS/VS
    {  // K tile [key][dk]: straight copy -> async path
      int row = t >> 2, cb = t & 3;
      size_t g = (size_t)(j0 + row) * 512 + h * 64 + cb * 16;
      async_cp16(Kp + g,     KSs + row * 72 + cb * 16);
      async_cp16(Kp + g + 8, KSs + row * 72 + cb * 16 + 8);
    }
    #pragma unroll
    for (int it = 0; it < 6; ++it) {  // V tile -> transposed [dv][key] (manual)
      int id = it * 256 + t;
      int key = id / 24, gb = id % 24, dv = gb * 8;
      H8 tv;
      tv.q = *(const v4u*)(Vp + (size_t)(j0 + key) * DIMD + h * DVV + dv);
      #pragma unroll
      for (int u = 0; u < 8; ++u) VSs[(dv + u) * 72 + key] = tv.h[u];
    }
    const int base = j0 - i0 + (NSEQ - 64);  // rel band start, in [0, 8064]
    #pragma unroll
    for (int it = 0; it < 4; ++it) {  // 128-row rel band [t][dk] -> async path
      int id = it * 256 + t;
      int r = id >> 3, gb = id & 7;
      async_cp16(Rk + ((size_t)h * NPOS + base + r) * DKK + gb * 8,
                 RSs + r * 72 + gb * 8);
    }
    async_wait0();
    __syncthreads();

    {  // content logits: S = Qc @ K^T (64x64), DK=64 -> 2 wmma steps
      v8f s0 = zero8, s1 = zero8;
      #pragma unroll
      for (int kc = 0; kc < 64; kc += 32) {
        v16h a  = ld_frag_a(QCs, 72, rt * 16 + ln, kc, hi);
        v16h b0 = ld_frag_b(KSs, 72, cw * 32 + ln, kc, hi);
        v16h b1 = ld_frag_b(KSs, 72, cw * 32 + 16 + ln, kc, hi);
        s0 = wmma16(a, b0, s0);
        s1 = wmma16(a, b1, s1);
      }
      #pragma unroll
      for (int r = 0; r < 8; ++r) {
        int row = rt * 16 + r + 8 * hi;
        Sf[row * 68 + cw * 32 + ln]      = s0[r];
        Sf[row * 68 + cw * 32 + 16 + ln] = s1[r];
      }
    }
    {  // band rel logits: S2 = Qp @ rel_band^T (64 x 128)
      v8f rc[4];
      #pragma unroll
      for (int i = 0; i < 4; ++i) rc[i] = zero8;
      #pragma unroll
      for (int kc = 0; kc < 64; kc += 32) {
        v16h a = ld_frag_a(QPs, 72, rt * 16 + ln, kc, hi);
        #pragma unroll
        for (int ct = 0; ct < 4; ++ct) {
          v16h b = ld_frag_b(RSs, 72, cw * 64 + ct * 16 + ln, kc, hi);
          rc[ct] = wmma16(a, b, rc[ct]);
        }
      }
      #pragma unroll
      for (int ct = 0; ct < 4; ++ct)
        #pragma unroll
        for (int r = 0; r < 8; ++r)
          S2[(rt * 16 + r + 8 * hi) * 132 + cw * 64 + ct * 16 + ln] = rc[ct][r];
    }
    __syncthreads();

    {  // fuse shifted rel (t = jj-ii+63) + online-softmax row max
      int row = t >> 2, qd = t & 3;
      float mx = -3.0e38f;
      #pragma unroll
      for (int c = 0; c < 16; ++c) {
        int col = qd * 16 + c;
        float sv = Sf[row * 68 + col] + S2[row * 132 + col - row + 63];
        Sf[row * 68 + col] = sv;
        mx = fmaxf(mx, sv);
      }
      mx = fmaxf(mx, __shfl_xor(mx, 1));
      mx = fmaxf(mx, __shfl_xor(mx, 2));
      if (qd == 0) {
        float mo = mrow[row];
        float mn = fmaxf(mo, mx);
        arow[row] = __expf(mo - mn);
        mrow[row] = mn;
      }
    }
    __syncthreads();
    {  // P = exp(S - m); l update
      int row = t >> 2, qd = t & 3;
      float mn = mrow[row], sum = 0.f;
      #pragma unroll
      for (int c = 0; c < 16; ++c) {
        int col = qd * 16 + c;
        float p = __expf(Sf[row * 68 + col] - mn);
        PSs[row * 72 + col] = (_Float16)p;
        sum += p;
      }
      sum += __shfl_xor(sum, 1);
      sum += __shfl_xor(sum, 2);
      if (qd == 0) lrow[row] = lrow[row] * arow[row] + sum;
    }
    __syncthreads();
    {  // O *= alpha; O += P @ V
      #pragma unroll
      for (int ct = 0; ct < 6; ++ct)
        #pragma unroll
        for (int r = 0; r < 8; ++r) o[ct][r] *= arow[rt * 16 + r + 8 * hi];
      #pragma unroll
      for (int kc = 0; kc < 64; kc += 32) {
        v16h a = ld_frag_a(PSs, 72, rt * 16 + ln, kc, hi);
        #pragma unroll
        for (int ct = 0; ct < 6; ++ct) {
          v16h b = ld_frag_b(VSs, 72, cw * 96 + ct * 16 + ln, kc, hi);
          o[ct] = wmma16(a, b, o[ct]);
        }
      }
    }
  }
  __syncthreads();
  #pragma unroll
  for (int ct = 0; ct < 6; ++ct)
    #pragma unroll
    for (int r = 0; r < 8; ++r) {
      int row = rt * 16 + r + 8 * hi;
      int col = cw * 96 + ct * 16 + ln;
      Oo[(size_t)(i0 + row) * DIMD + h * DVV + col] =
          (_Float16)(o[ct][r] / lrow[row]);
    }
}

// ---------------- host orchestration ----------------
extern "C" void kernel_launch(void* const* d_in, const int* in_sizes, int n_in,
                              void* d_out, int out_size, void* d_ws, size_t ws_size,
                              hipStream_t stream) {
  const float* x  = (const float*)d_in[0];
  const float* g  = (const float*)d_in[1];
  const float* be = (const float*)d_in[2];
  const float* Wq = (const float*)d_in[3];
  const float* Wk = (const float*)d_in[4];
  const float* Wv = (const float*)d_in[5];
  const float* Wr = (const float*)d_in[6];
  const float* Wo = (const float*)d_in[7];
  const float* bo = (const float*)d_in[8];
  const float* cb = (const float*)d_in[9];
  const float* pb = (const float*)d_in[10];
  float* out = (float*)d_out;
  (void)in_sizes; (void)n_in; (void)out_size; (void)ws_size;

  size_t off = 0;
  auto carve = [&](size_t bytes) -> void* {
    void* p = (char*)d_ws + off;
    off += (bytes + 255) & ~(size_t)255;
    return p;
  };
  _Float16* xn16 = (_Float16*)carve((size_t)NSEQ * DIMD * 2);
  _Float16* Wq16 = (_Float16*)carve((size_t)DIMD * 512 * 2);
  _Float16* Wk16 = (_Float16*)carve((size_t)DIMD * 512 * 2);
  _Float16* Wv16 = (_Float16*)carve((size_t)DIMD * DIMD * 2);
  _Float16* Wo16 = (_Float16*)carve((size_t)DIMD * DIMD * 2);
  _Float16* RK16 = (_Float16*)carve((size_t)NH * NPOS * DKK * 2);
  float*    Qf   = (float*)   carve((size_t)NSEQ * 512 * 4);
  _Float16* Qc16 = (_Float16*)carve((size_t)NSEQ * 512 * 2);
  _Float16* Qp16 = (_Float16*)carve((size_t)NSEQ * 512 * 2);
  _Float16* K16  = (_Float16*)carve((size_t)NSEQ * 512 * 2);
  _Float16* V16  = (_Float16*)carve((size_t)NSEQ * DIMD * 2);
  _Float16* O16  = (_Float16*)carve((size_t)NSEQ * DIMD * 2);

  // 1) LayerNorm + casts + rel_k basis
  ln_k<<<NSEQ, 256, 0, stream>>>(x, g, be, xn16);
  cast_k<<<(DIMD * 512 + 255) / 256, 256, 0, stream>>>(Wq, Wq16, DIMD * 512);
  cast_k<<<(DIMD * 512 + 255) / 256, 256, 0, stream>>>(Wk, Wk16, DIMD * 512);
  cast_k<<<(DIMD * DIMD + 255) / 256, 256, 0, stream>>>(Wv, Wv16, DIMD * DIMD);
  cast_k<<<(DIMD * DIMD + 255) / 256, 256, 0, stream>>>(Wo, Wo16, DIMD * DIMD);
  relk_k<<<NPOS, 128, 0, stream>>>(Wr, RK16);

  // 2) projections (WMMA GEMMs)
  gemm_k<1><<<dim3(NSEQ / 64, 512 / 64), 256, 0, stream>>>(
      xn16, Wq16, 512, DIMD, nullptr, Qf, nullptr);
  gemm_k<0><<<dim3(NSEQ / 64, 512 / 64), 256, 0, stream>>>(
      xn16, Wk16, 512, DIMD, K16, nullptr, nullptr);
  gemm_k<0><<<dim3(NSEQ / 64, DIMD / 64), 256, 0, stream>>>(
      xn16, Wv16, DIMD, DIMD, V16, nullptr, nullptr);

  // 3) Q scale + biases
  qbias_k<<<(NSEQ * 512) / 256, 256, 0, stream>>>(Qf, cb, pb, Qc16, Qp16);

  // 4) flash attention with rel-shift band
  attn_k<<<dim3(NSEQ / 64, NH), 256, AT_SMEM, stream>>>(Qc16, Qp16, K16, V16,
                                                        RK16, O16);

  // 5) output projection + bias
  gemm_k<1><<<dim3(NSEQ / 64, DIMD / 64), 256, 0, stream>>>(
      O16, Wo16, DIMD, DIMD, nullptr, out, bo);
}
